// MultiHeadAttention_58909771432575
// MI455X (gfx1250) — matrix operand order repaired and checked by reading, not compiled, not run on hardware
//
#include <hip/hip_runtime.h>

// ---------------------------------------------------------------------------
// Problem constants (match reference)
// ---------------------------------------------------------------------------
#define BB 2
#define SS 2048
#define DD 1024
#define HH 16
#define HD 64
#define MM (BB * SS)   // 4096 rows in the x@W^T GEMMs
#define KK DD          // 1024
#define NN DD          // 1024

typedef __attribute__((ext_vector_type(16))) __bf16 v16bf;
typedef __attribute__((ext_vector_type(8)))  __bf16 v8bf;
typedef __attribute__((ext_vector_type(4)))  __bf16 v4bf;
typedef __attribute__((ext_vector_type(4)))  float  v4f;
typedef __attribute__((ext_vector_type(8)))  float  v8f;
typedef __attribute__((ext_vector_type(4)))  int    v4i;

#define AS1 __attribute__((address_space(1)))
#define AS3 __attribute__((address_space(3)))

// hardware transcendental: v_exp_f32 computes 2^x
__device__ __forceinline__ float hw_exp2(float x) {
    return __builtin_amdgcn_exp2f(x);
}

// ---------------------------------------------------------------------------
// Helpers
// ---------------------------------------------------------------------------
__device__ inline v8f zero8() {
    v8f z;
#pragma unroll
    for (int i = 0; i < 8; ++i) z[i] = 0.0f;
    return z;
}

// A-matrix fragment (16x32 bf16, MxK). lane&15 = row M. lanes<16 hold
// K = {k0..k0+7, k0+16..k0+23}; lanes>=16 hold K = {k0+8..k0+15, k0+24..k0+31}.
__device__ __forceinline__ v16bf load_a_frag(const __bf16* row, int k0, int hi) {
    v8bf x = *(const v8bf*)(row + k0 + hi * 8);
    v8bf y = *(const v8bf*)(row + k0 + 16 + hi * 8);
    v16bf a;
#pragma unroll
    for (int i = 0; i < 8; ++i) { a[i] = x[i]; a[i + 8] = y[i]; }
    return a;
}

// B-matrix fragment (32x16 bf16, KxN). lane&15 = col N. lanes<16 hold
// K = {k0..k0+7, k0+16..k0+23}; lanes>=16 hold K = {k0+8..k0+15, k0+24..k0+31}.
__device__ __forceinline__ v16bf load_b_frag(const __bf16* row, int k0, int hi) {
    v8bf x = *(const v8bf*)(row + k0 + hi * 8);
    v8bf y = *(const v8bf*)(row + k0 + 16 + hi * 8);
    v16bf b;
#pragma unroll
    for (int i = 0; i < 8; ++i) { b[i] = x[i]; b[i + 8] = y[i]; }
    return b;
}

__device__ __forceinline__ v8f wmma_bf16(v16bf a, v16bf b, v8f c) {
    return __builtin_amdgcn_wmma_f32_16x16x32_bf16(
        false, a, false, b, (short)0, c, false, false);
}

// In-wave LDS store->load visibility (wave-private buffer, cross-lane exchange)
__device__ __forceinline__ void lds_sync() {
    __builtin_amdgcn_wave_barrier();
    asm volatile("s_wait_dscnt 0" ::: "memory");
    __builtin_amdgcn_wave_barrier();
}

// 16B global -> LDS async copy (GLOBAL_LOAD_ASYNC_TO_LDS_B128, ASYNCcnt)
__device__ __forceinline__ void async_copy_b128(const __bf16* g, __bf16* l) {
#if __has_builtin(__builtin_amdgcn_global_load_async_to_lds_b128)
    __builtin_amdgcn_global_load_async_to_lds_b128(
        (AS1 v4i*)(AS1 void*)g, (AS3 v4i*)(AS3 void*)l, 0, 0);
#else
    *(v8bf*)l = *(const v8bf*)g;   // fallback: register bounce
#endif
}

__device__ __forceinline__ void wait_async() {
#if __has_builtin(__builtin_amdgcn_s_wait_asynccnt)
    __builtin_amdgcn_s_wait_asynccnt(0);
#else
    asm volatile("s_wait_asynccnt 0" ::: "memory");
#endif
}

// ---------------------------------------------------------------------------
// Kernel 0: f32 -> bf16 conversion (vectorized 4-wide)
// ---------------------------------------------------------------------------
__global__ void __launch_bounds__(256) cvt_bf16_kernel(
        const v4f* __restrict__ in, v4bf* __restrict__ out, int n4) {
    int idx = blockIdx.x * blockDim.x + threadIdx.x;
    int stride = gridDim.x * blockDim.x;
    for (int i = idx; i < n4; i += stride) {
        v4f x = in[i];
        v4bf y;
#pragma unroll
        for (int j = 0; j < 4; ++j) y[j] = (__bf16)x[j];
        out[i] = y;
    }
}

// ---------------------------------------------------------------------------
// GEMM: C[m,n] = sum_k A[m,k] * W[n,k]   (x @ W^T)
// Each wave computes a 32x64 output tile: 8 WMMA accumulators, A reused 4x.
// MODE 0: out bf16, head-major [B,H,S,HD]            (Q, K projections)
// MODE 2: out bf16, head-major transposed [B,H,HD,S] (V projection)
// MODE 3: out f32, plain row-major [M,N]             (final @ wo^T)
// ---------------------------------------------------------------------------
template <int MODE>
__global__ void __launch_bounds__(128, 1) gemm_xwT_kernel(
        const __bf16* __restrict__ A, const __bf16* __restrict__ W,
        __bf16* __restrict__ outB, float* __restrict__ outF) {
    const int lane = threadIdx.x & 31;
    const int l16  = lane & 15;
    const int hi   = lane >> 4;
    const int wid  = blockIdx.x * (blockDim.x >> 5) + (threadIdx.x >> 5);
    const int mt   = wid >> 4;        // 0..127  (MM/32 tiles)
    const int nt   = wid & 15;        // 0..15   (NN/64 tiles)
    const int m0   = mt * 32;
    const int n0   = nt * 64;

    const __bf16* arow0 = A + (size_t)(m0 + l16) * KK;
    const __bf16* arow1 = A + (size_t)(m0 + 16 + l16) * KK;
    const __bf16* wrow[4];
#pragma unroll
    for (int ni = 0; ni < 4; ++ni)
        wrow[ni] = W + (size_t)(n0 + ni * 16 + l16) * KK;

    v8f acc[2][4];
#pragma unroll
    for (int mi = 0; mi < 2; ++mi)
#pragma unroll
        for (int ni = 0; ni < 4; ++ni) acc[mi][ni] = zero8();

    for (int k0 = 0; k0 < KK; k0 += 64) {
        // keep WGP$ warm 4 k-steps ahead (global_prefetch_b8; speculative,
        // OOB translations silently dropped)
        __builtin_prefetch((const void*)(arow0 + k0 + 256), 0, 3);
        __builtin_prefetch((const void*)(arow1 + k0 + 256), 0, 3);
#pragma unroll
        for (int ni = 0; ni < 4; ++ni)
            __builtin_prefetch((const void*)(wrow[ni] + k0 + 256), 0, 3);
#pragma unroll
        for (int kk = 0; kk < 64; kk += 32) {
            const int k = k0 + kk;
            v16bf a0 = load_a_frag(arow0, k, hi);
            v16bf a1 = load_a_frag(arow1, k, hi);
#pragma unroll
            for (int ni = 0; ni < 4; ++ni) {
                v16bf b = load_b_frag(wrow[ni], k, hi);
                acc[0][ni] = wmma_bf16(a0, b, acc[0][ni]);
                acc[1][ni] = wmma_bf16(a1, b, acc[1][ni]);
            }
        }
    }

    // store: C element (vgpr i) is row = base + i + 8*hi, col = base + l16
#pragma unroll
    for (int mi = 0; mi < 2; ++mi) {
#pragma unroll
        for (int ni = 0; ni < 4; ++ni) {
#pragma unroll
            for (int i = 0; i < 8; ++i) {
                int m = m0 + mi * 16 + i + hi * 8;
                int n = n0 + ni * 16 + l16;
                float val = acc[mi][ni][i];
                if (MODE == 3) {
                    outF[(size_t)m * NN + n] = val;
                } else {
                    int b  = m >> 11;       // m / SS
                    int s  = m & (SS - 1);
                    int h  = n >> 6;        // n / HD
                    int hd = n & (HD - 1);
                    size_t off;
                    if (MODE == 0)
                        off = (((size_t)(b * HH + h)) * SS + s) * HD + hd;
                    else
                        off = (((size_t)(b * HH + h)) * HD + hd) * SS + s;
                    outB[off] = (__bf16)val;
                }
            }
        }
    }
}

// ---------------------------------------------------------------------------
// Attention chunk (64 keys, tiles already staged in LDS).
// Scores (8 WMMA), online softmax (exp2 domain, row-sum as P@ones on the
// matrix pipe), P through wave-private LDS (C->A layout), P*V (8 WMMA).
// MASK=true only for the single diagonal-straddling chunk.
// Kt: [64 keys][64 hd] bf16 in LDS.  Vt_l: [64 hd][64 keys] bf16 in LDS.
// ---------------------------------------------------------------------------
template <bool MASK>
__device__ __forceinline__ void attn_chunk(
        int kb, int q0, int l16, int hi,
        const __bf16* Kt, const __bf16* Vt_l,
        __bf16* pb, const v16bf& qa0, const v16bf& qa1, const v16bf& ones,
        v8f& o0, v8f& o1, v8f& o2, v8f& o3, v8f& lacc,
        float (&mrow)[8], float scale) {
    // ---- scores: 16x64 tile as four 16x16 WMMA outputs (K from LDS) -------
    v8f s[4];
#pragma unroll
    for (int kg = 0; kg < 4; ++kg) {
        const __bf16* krow = Kt + (kg * 16 + l16) * HD;
        v8f t = zero8();
        t = wmma_bf16(qa0, load_b_frag(krow, 0, hi), t);
        t = wmma_bf16(qa1, load_b_frag(krow, 32, hi), t);
        s[kg] = t;
    }

    // ---- scale (+ causal mask) + row max ----------------------------------
    float mx[8];
#pragma unroll
    for (int i = 0; i < 8; ++i) {
        float e0 = s[0][i] * scale;
        float e1 = s[1][i] * scale;
        float e2 = s[2][i] * scale;
        float e3 = s[3][i] * scale;
        if (MASK) {
            int qidx = q0 + i + hi * 8;
            if (kb + l16 > qidx)      e0 = -1e30f;
            if (kb + 16 + l16 > qidx) e1 = -1e30f;
            if (kb + 32 + l16 > qidx) e2 = -1e30f;
            if (kb + 48 + l16 > qidx) e3 = -1e30f;
        }
        s[0][i] = e0; s[1][i] = e1; s[2][i] = e2; s[3][i] = e3;
        mx[i] = fmaxf(fmaxf(e0, e1), fmaxf(e2, e3));
    }
#pragma unroll
    for (int m = 1; m <= 8; m <<= 1)
#pragma unroll
        for (int i = 0; i < 8; ++i)
            mx[i] = fmaxf(mx[i], __shfl_xor(mx[i], m, 32));

    // ---- online softmax update (exp2 domain); no shuffle row-sum ----------
#pragma unroll
    for (int i = 0; i < 8; ++i) {
        float mnew  = fmaxf(mrow[i], mx[i]);
        float alpha = hw_exp2(mrow[i] - mnew);
        mrow[i] = mnew;
        o0[i] *= alpha; o1[i] *= alpha; o2[i] *= alpha; o3[i] *= alpha;
        lacc[i] *= alpha;
        s[0][i] = hw_exp2(s[0][i] - mnew);
        s[1][i] = hw_exp2(s[1][i] - mnew);
        s[2][i] = hw_exp2(s[2][i] - mnew);
        s[3][i] = hw_exp2(s[3][i] - mnew);
    }

    // ---- P: C-layout -> A-layout via wave-private LDS (16x64 tile) --------
#pragma unroll
    for (int i = 0; i < 8; ++i) {
        int r = i + hi * 8;
        pb[r * 64 + l16]      = (__bf16)s[0][i];
        pb[r * 64 + 16 + l16] = (__bf16)s[1][i];
        pb[r * 64 + 32 + l16] = (__bf16)s[2][i];
        pb[r * 64 + 48 + l16] = (__bf16)s[3][i];
    }
    lds_sync();
    v16bf pa0 = load_a_frag(pb + l16 * 64, 0, hi);   // local keys 0..31
    v16bf pa1 = load_a_frag(pb + l16 * 64, 32, hi);  // local keys 32..63

    // ---- row-sum on the matrix pipe: lacc += P @ ones ---------------------
    lacc = wmma_bf16(pa0, ones, lacc);
    lacc = wmma_bf16(pa1, ones, lacc);

    // ---- O += P * V (V tile rows contiguous along keys, from LDS) ---------
    const __bf16* v0 = Vt_l + (0 * 16 + l16) * 64;
    const __bf16* v1 = Vt_l + (1 * 16 + l16) * 64;
    const __bf16* v2 = Vt_l + (2 * 16 + l16) * 64;
    const __bf16* v3 = Vt_l + (3 * 16 + l16) * 64;
    o0 = wmma_bf16(pa0, load_b_frag(v0, 0, hi), o0);
    o0 = wmma_bf16(pa1, load_b_frag(v0, 32, hi), o0);
    o1 = wmma_bf16(pa0, load_b_frag(v1, 0, hi), o1);
    o1 = wmma_bf16(pa1, load_b_frag(v1, 32, hi), o1);
    o2 = wmma_bf16(pa0, load_b_frag(v2, 0, hi), o2);
    o2 = wmma_bf16(pa1, load_b_frag(v2, 32, hi), o2);
    o3 = wmma_bf16(pa0, load_b_frag(v3, 0, hi), o3);
    o3 = wmma_bf16(pa1, load_b_frag(v3, 32, hi), o3);
    lds_sync();
}

// ---------------------------------------------------------------------------
// Kernel 4: causal flash-attention with block-cooperative K/V staging.
// Block = 4 waves = 4 consecutive 16-query blocks of ONE head, so every wave
// in the block has the IDENTICAL chunk count (nchunks = qgroup+1): barriers
// are uniform. K (8KB, contiguous) and V (8KB) tiles staged into LDS with
// async global->LDS copies, shared by all 4 waves (4x less L2 traffic).
// Qh/Kh: [B,H,S,HD] bf16.  Vt: [B,H,HD,S] bf16.  Ctx: [B*S, D] bf16.
// ---------------------------------------------------------------------------
__global__ void __launch_bounds__(128, 1) attn_kernel(
        const __bf16* __restrict__ Qh, const __bf16* __restrict__ Kh,
        const __bf16* __restrict__ Vt, __bf16* __restrict__ Ctx) {
    __shared__ __bf16 Kt[64 * 64];        // [key][hd]   8 KB
    __shared__ __bf16 Vtile[64 * 64];     // [hd][key]   8 KB
    __shared__ __bf16 pbuf[4][16 * 64];   // per-wave P tiles, 8 KB

    const int tid   = threadIdx.x;
    const int lane  = tid & 31;
    const int l16   = lane & 15;
    const int hi    = lane >> 4;
    const int wslot = tid >> 5;
    const int bh    = blockIdx.x >> 5;    // b*H + h
    const int qg    = blockIdx.x & 31;    // query group (64 queries)
    const int qb    = qg * 4 + wslot;     // this wave's 16-query block
    const int q0    = qb * 16;
    const int nchunks = qg + 1;           // identical for all 4 waves

    const __bf16* Qbase = Qh + (size_t)bh * SS * HD;
    const __bf16* Kbase = Kh + (size_t)bh * SS * HD;
    const __bf16* Vbase = Vt + (size_t)bh * HD * SS;

    // Q A-fragments: 16 rows x 64, reused all iterations
    const __bf16* qrow = Qbase + (size_t)(q0 + l16) * HD;
    const v16bf qa0 = load_a_frag(qrow, 0, hi);
    const v16bf qa1 = load_a_frag(qrow, 32, hi);

    v16bf ones;
#pragma unroll
    for (int i = 0; i < 16; ++i) ones[i] = (__bf16)1.0f;

    v8f o0 = zero8(), o1 = zero8(), o2 = zero8(), o3 = zero8();
    v8f lacc = zero8();
    float mrow[8];
#pragma unroll
    for (int i = 0; i < 8; ++i) mrow[i] = -1e30f;

    // fold 1/sqrt(HD) and log2(e) into the score scale; use exp2 throughout
    const float scale = 0.125f * 1.44269504088896340736f;
    __bf16* pb = &pbuf[wslot][0];

    for (int kc = 0; kc < nchunks; ++kc) {
        const int kb = kc * 64;

        __syncthreads();   // previous chunk's consumers done with tiles

        // ---- stage K tile: 64 rows x 64 hd = one contiguous 8KB region ----
        {
            const __bf16* ksrc = Kbase + (size_t)kb * HD;
#pragma unroll
            for (int p = 0; p < 4; ++p) {
                int off = (p * 128 + tid) * 8;   // 16B granules
                async_copy_b128(ksrc + off, &Kt[off]);
            }
            // ---- stage V tile: 64 hd-rows x 64 keys (row stride SS) -------
#pragma unroll
            for (int p = 0; p < 4; ++p) {
                int idx   = p * 128 + tid;       // 0..511
                int row   = idx >> 3;            // hd row 0..63
                int piece = (idx & 7) * 8;       // 16B granule in row
                async_copy_b128(Vbase + (size_t)row * SS + kb + piece,
                                &Vtile[row * 64 + piece]);
            }
        }
        wait_async();        // own ASYNCcnt drained
        __syncthreads();     // all waves' copies visible

        // prefetch next chunk's K/V sources while computing this one
        __builtin_prefetch((const void*)(Kbase + (size_t)(kb + 64) * HD + tid * 32), 0, 3);
        __builtin_prefetch((const void*)(Vbase + (size_t)(tid >> 1) * SS + kb + 64 + (tid & 1) * 32), 0, 3);

        if (kc == nchunks - 1)
            attn_chunk<true>(kb, q0, l16, hi, Kt, Vtile, pb, qa0, qa1, ones,
                             o0, o1, o2, o3, lacc, mrow, scale);
        else
            attn_chunk<false>(kb, q0, l16, hi, Kt, Vtile, pb, qa0, qa1, ones,
                              o0, o1, o2, o3, lacc, mrow, scale);
    }

    // ---- normalize + store context [B*S, D] bf16 --------------------------
    const int b = bh >> 4;
    const int h = bh & 15;
#pragma unroll
    for (int i = 0; i < 8; ++i) {
        float inv = 1.0f / lacc[i];
        int srow = q0 + i + hi * 8;
        size_t base = ((size_t)b * SS + srow) * DD + h * HD + l16;
        Ctx[base + 0]  = (__bf16)(o0[i] * inv);
        Ctx[base + 16] = (__bf16)(o1[i] * inv);
        Ctx[base + 32] = (__bf16)(o2[i] * inv);
        Ctx[base + 48] = (__bf16)(o3[i] * inv);
    }
}

// ---------------------------------------------------------------------------
// Launch
// ---------------------------------------------------------------------------
extern "C" void kernel_launch(void* const* d_in, const int* in_sizes, int n_in,
                              void* d_out, int out_size, void* d_ws, size_t ws_size,
                              hipStream_t stream) {
    const float* q  = (const float*)d_in[0];
    const float* k  = (const float*)d_in[1];
    const float* v  = (const float*)d_in[2];
    const float* wq = (const float*)d_in[3];
    const float* wk = (const float*)d_in[4];
    const float* wv = (const float*)d_in[5];
    const float* wo = (const float*)d_in[6];
    float* out = (float*)d_out;

    const size_t nx = (size_t)BB * SS * DD;   // 4,194,304
    const size_t nw = (size_t)DD * DD;        // 1,048,576

    __bf16* ws  = (__bf16*)d_ws;
    __bf16* qb  = ws;
    __bf16* kb  = qb + nx;
    __bf16* vb  = kb + nx;
    __bf16* wqb = vb + nx;
    __bf16* wkb = wqb + nw;
    __bf16* wvb = wkb + nw;
    __bf16* wob = wvb + nw;
    __bf16* Qh  = wob + nw;
    __bf16* Kh  = Qh + nx;
    __bf16* Vt  = Kh + nx;
    __bf16* Ctx = Vt + nx;

    // 0) one-time f32 -> bf16 conversions (4-wide vectorized)
    cvt_bf16_kernel<<<1024, 256, 0, stream>>>((const v4f*)q,  (v4bf*)qb,  (int)(nx / 4));
    cvt_bf16_kernel<<<1024, 256, 0, stream>>>((const v4f*)k,  (v4bf*)kb,  (int)(nx / 4));
    cvt_bf16_kernel<<<1024, 256, 0, stream>>>((const v4f*)v,  (v4bf*)vb,  (int)(nx / 4));
    cvt_bf16_kernel<<<256,  256, 0, stream>>>((const v4f*)wq, (v4bf*)wqb, (int)(nw / 4));
    cvt_bf16_kernel<<<256,  256, 0, stream>>>((const v4f*)wk, (v4bf*)wkb, (int)(nw / 4));
    cvt_bf16_kernel<<<256,  256, 0, stream>>>((const v4f*)wv, (v4bf*)wvb, (int)(nw / 4));
    cvt_bf16_kernel<<<256,  256, 0, stream>>>((const v4f*)wo, (v4bf*)wob, (int)(nw / 4));

    // 1-3) QKV projections (WMMA, 32x64 wave tiles): 2048 tiles, 4 waves/block
    gemm_xwT_kernel<0><<<512, 128, 0, stream>>>(qb, wqb, Qh, nullptr);
    gemm_xwT_kernel<0><<<512, 128, 0, stream>>>(kb, wkb, Kh, nullptr);
    gemm_xwT_kernel<2><<<512, 128, 0, stream>>>(vb, wvb, Vt, nullptr);

    // 4) causal flash-attention: 1024 blocks x (4 waves sharing K/V tiles)
    attn_kernel<<<1024, 128, 0, stream>>>(Qh, Kh, Vt, Ctx);

    // 5) output projection (WMMA, f32 store)
    gemm_xwT_kernel<3><<<512, 128, 0, stream>>>(Ctx, wob, nullptr, out);
}
